// SpixelNet_geom_24919400251933
// MI455X (gfx1250) — matrix operand order repaired and checked by reading, not compiled, not run on hardware
//
#include <hip/hip_runtime.h>
#include <hip/hip_bf16.h>

typedef _Float16 f16;
typedef __attribute__((ext_vector_type(16))) _Float16 v16h;
typedef __attribute__((ext_vector_type(8)))  _Float16 v8h;
typedef __attribute__((ext_vector_type(8)))  float    v8f;

#define BB 16
#define HH 512
#define WW 512

// ---------------- async global->LDS probe ----------------
#ifndef __has_builtin
#define __has_builtin(x) 0
#endif

#if __has_builtin(__builtin_amdgcn_global_load_async_to_lds_b128)
#define HAS_ASYNC 1
// probe-derived signature: (as1 int4* global, as3 int4* lds, int offset, int cpol)
typedef int async_v4i __attribute__((vector_size(16)));
#define GPTR(p) ((__attribute__((address_space(1))) async_v4i*)(unsigned long long)(const void*)(p))
#define LPTR(p) ((__attribute__((address_space(3))) async_v4i*)(unsigned)(unsigned long long)(const void*)(p))
#define ASYNC_CP_B128(g, l) __builtin_amdgcn_global_load_async_to_lds_b128(GPTR(g), LPTR(l), 0, 0)
#if __has_builtin(__builtin_amdgcn_s_wait_asynccnt)
#define ASYNC_WAIT() __builtin_amdgcn_s_wait_asynccnt(0)
#else
#define ASYNC_WAIT() asm volatile("s_wait_asynccnt 0" ::: "memory")
#endif
#else
#define HAS_ASYNC 0
#endif

// ---------------- workspace layout (bytes) ----------------
#define WS_EDGE   0ull                                   // 16*512*512*4      = 16777216
#define WS_STATS  16777216ull                            // 8 layers * 512 f  = 16384
#define WS_SCAL   (WS_STATS + 16384ull)                  // asum, esum        = 256
#define WS_CONVED (WS_SCAL + 256ull)                     // 16*32*32*2*4      = 131072
#define WS_ACTA   (WS_CONVED + 131072ull)                // 16*512*512*16*2   = 134217728
#define WS_ACTB   (WS_ACTA + 134217728ull)               // 16*256*256*16*2   = 33554432

__device__ __forceinline__ float diff_round(float x) {
    return x - __sinf(6.28318530718f * x) * 0.15915494309f;
}
__device__ __forceinline__ float gelu_tanh(float x) {
    float x3 = x * x * x;
    return 0.5f * x * (1.f + tanhf(0.7978845608f * (x + 0.044715f * x3)));
}
__device__ __forceinline__ float wred(float v) {
#pragma unroll
    for (int off = 16; off >= 1; off >>= 1) v += __shfl_xor(v, off, 32);
    return v;
}

// ---------------- zero small scratch ----------------
__global__ void k_zero(float* p, int n) {
    int i = blockIdx.x * 256 + threadIdx.x;
    if (i < n) p[i] = 0.f;
}

// ---------------- Farid edge magnitude ----------------
__global__ void k_farid(const float* __restrict__ img, float* __restrict__ edge) {
    int idx = blockIdx.x * 256 + threadIdx.x;         // 16*512*512
    int x = idx & 511, y = (idx >> 9) & 511, b = idx >> 18;
    const float P[5] = {0.03032f, 0.249724f, 0.439911f, 0.249724f, 0.03032f};
    const float D[5] = {0.10455f, 0.292315f, 0.0f, -0.292315f, -0.10455f};
    const float* im = img + (size_t)b * HH * WW;
    float gx = 0.f, gy = 0.f;
#pragma unroll
    for (int i = 0; i < 5; i++) {
        int iy = y + i - 2;
        if (iy < 0 || iy >= HH) continue;
#pragma unroll
        for (int j = 0; j < 5; j++) {
            int ix = x + j - 2;
            if (ix < 0 || ix >= WW) continue;
            float v = im[iy * WW + ix];
            gx += P[i] * D[j] * v;
            gy += D[i] * P[j] * v;
        }
    }
    edge[idx] = sqrtf(gx * gx + gy * gy + 1e-8f);
}

// ---------------- conv0: 3x3, 2->16 ch, stride 1, WMMA (K=18 pad 32) ----------------
__global__ void k_conv0(const float* __restrict__ img, const float* __restrict__ edge,
                        const float* __restrict__ wfirst, const float* __restrict__ bfirst,
                        f16* __restrict__ out) {
    __shared__ __align__(32) f16 ldsA[8][16][32];
    __shared__ __align__(32) f16 ldsW[16][32];        // [n][k] : B-fragment contiguous
    int t = threadIdx.x, wave = t >> 5, lane = t & 31;

#if HAS_ASYNC
    __shared__ __align__(16) float ldsWstage[288];
    for (int i = t; i < 72; i += 256)                 // 288 f32 = 72 x b128
        ASYNC_CP_B128(wfirst + i * 4, &ldsWstage[i * 4]);
#endif

    int tile = blockIdx.x * 8 + wave;                 // b*(512*32) + y*32 + xt
    int xt = tile & 31, y = (tile >> 5) & 511, b = tile >> 14;
    int x0 = xt << 4;
    for (int i = lane; i < 16 * 32; i += 32) {
        int p = i >> 5, k = i & 31;
        float v = 0.f;
        if (k < 18) {
            int tap = k >> 1, ci = k & 1;
            int iy = y + tap / 3 - 1, ix = x0 + p + tap % 3 - 1;
            if (iy >= 0 && iy < HH && ix >= 0 && ix < WW) {
                size_t o = ((size_t)b * HH + iy) * WW + ix;
                v = ci ? edge[o] : img[o];
            }
        }
        ldsA[wave][p][k] = (f16)v;
    }

#if HAS_ASYNC
    ASYNC_WAIT();
    __syncthreads();
    for (int i = t; i < 16 * 32; i += 256) {
        int n = i >> 5, k = i & 31;
        ldsW[n][k] = (f16)(k < 18 ? ldsWstage[k * 16 + n] : 0.f);
    }
#else
    for (int i = t; i < 16 * 32; i += 256) {
        int n = i >> 5, k = i & 31;
        ldsW[n][k] = (f16)(k < 18 ? wfirst[k * 16 + n] : 0.f);
    }
#endif
    __syncthreads();

    int half_ = lane >> 4, m = lane & 15, n = lane & 15;
    // A fragment: k = 16*half + e, contiguous
    v8h alo = *(const v8h*)&ldsA[wave][m][half_ * 16];
    v8h ahi = *(const v8h*)&ldsA[wave][m][half_ * 16 + 8];
    v16h a = __builtin_shufflevector(alo, ahi, 0, 1, 2, 3, 4, 5, 6, 7,
                                     8, 9, 10, 11, 12, 13, 14, 15);
    // B fragment: k = 16*half + e, contiguous in [n][k] layout
    v16h bm = *(const v16h*)&ldsW[n][half_ * 16];
    v8f c = {};
    c = __builtin_amdgcn_wmma_f32_16x16x32_f16(false, a, false, bm, (short)0, c, false, false);
    float bias = bfirst[n];
#pragma unroll
    for (int r = 0; r < 8; r++) {
        int px = x0 + r + 8 * half_;
        out[(((size_t)b * HH + y) * WW + px) * 16 + n] = (f16)(c[r] + bias);
    }
}

// ---------------- mid conv: 3x3, 16->16, stride s, WMMA (K=144 pad 160) ----------------
__global__ void k_convmid(const f16* __restrict__ in, const float* __restrict__ w,
                          const float* __restrict__ bias, f16* __restrict__ out,
                          int Hi, int Wi, int Ho, int Wo, int stride) {
    __shared__ __align__(32) f16 ldsA[8][16][160];
    __shared__ __align__(32) f16 ldsW[16][160];       // [n][k] transposed for B fragments
    int t = threadIdx.x, wave = t >> 5, lane = t & 31;

#if HAS_ASYNC
    __shared__ __align__(16) float ldsWstage[2304];
    for (int i = t; i < 576; i += 256)                // 2304 f32 = 576 x b128
        ASYNC_CP_B128(w + i * 4, &ldsWstage[i * 4]);
#endif

    int tpr = Wo >> 4;
    int tile = blockIdx.x * 8 + wave;
    int xt = tile % tpr;
    int rem = tile / tpr;
    int y = rem % Ho, b = rem / Ho;
    int x0 = xt << 4;

    __builtin_prefetch(in + (((size_t)b * Hi + y * stride) * Wi + x0 * stride) * 16, 0, 1);

    // im2col gather: 144 vectors of 16 contiguous f16 (one (pixel, tap) each)
    for (int i = lane; i < 144; i += 32) {
        int p = i / 9, tap = i % 9;
        int iy = y * stride + tap / 3 - 1;
        int ix = (x0 + p) * stride + tap % 3 - 1;
        v16h v = {};
        if (iy >= 0 && iy < Hi && ix >= 0 && ix < Wi)
            v = *(const v16h*)(in + (((size_t)b * Hi + iy) * Wi + ix) * 16);
        *(v16h*)&ldsA[wave][p][tap * 16] = v;
    }
    if (lane < 16) { v16h z = {}; *(v16h*)&ldsA[wave][lane][144] = z; }  // K pad

#if HAS_ASYNC
    ASYNC_WAIT();
    __syncthreads();
    for (int i = t; i < 16 * 160; i += 256) {
        int n = i / 160, k = i % 160;
        ldsW[n][k] = (f16)(k < 144 ? ldsWstage[k * 16 + n] : 0.f);
    }
#else
    for (int i = t; i < 16 * 160; i += 256) {
        int n = i / 160, k = i % 160;
        ldsW[n][k] = (f16)(k < 144 ? w[k * 16 + n] : 0.f);
    }
#endif
    __syncthreads();

    int half_ = lane >> 4, m = lane & 15, n = lane & 15;
    v8f c = {};
#pragma unroll
    for (int ks = 0; ks < 5; ks++) {
        int kb0 = ks * 32;
        v8h alo = *(const v8h*)&ldsA[wave][m][kb0 + half_ * 16];
        v8h ahi = *(const v8h*)&ldsA[wave][m][kb0 + half_ * 16 + 8];
        v16h a = __builtin_shufflevector(alo, ahi, 0, 1, 2, 3, 4, 5, 6, 7,
                                         8, 9, 10, 11, 12, 13, 14, 15);
        v16h bm = *(const v16h*)&ldsW[n][kb0 + half_ * 16];
        c = __builtin_amdgcn_wmma_f32_16x16x32_f16(false, a, false, bm, (short)0, c, false, false);
    }
    float bv = bias[n];
#pragma unroll
    for (int r = 0; r < 8; r++) {
        int px = x0 + r + 8 * half_;
        out[(((size_t)b * Ho + y) * Wo + px) * 16 + n] = (f16)(c[r] + bv);
    }
}

// ---------------- instance-norm stats (sum, sumsq per (b, c)) ----------------
__global__ void k_stats(const f16* __restrict__ act, float* __restrict__ stats,
                        int npix, int nchunks) {
    __shared__ float rs[256], rq[256];
    int b = blockIdx.x / nchunks, chunk = blockIdx.x % nchunks;
    int t = threadIdx.x, c = t & 15, g = t >> 4;
    int P = (npix + nchunks - 1) / nchunks;
    int p0 = chunk * P;
    int p1 = p0 + P; if (p1 > npix) p1 = npix;
    const f16* base = act + (size_t)b * npix * 16;
    float s = 0.f, q = 0.f;
    for (int p = p0 + g; p < p1; p += 16) {
        float v = (float)base[(size_t)p * 16 + c];
        s += v; q += v * v;
    }
    rs[t] = s; rq[t] = q;
    __syncthreads();
    for (int off = 128; off >= 16; off >>= 1) {
        if (t < off) { rs[t] += rs[t + off]; rq[t] += rq[t + off]; }
        __syncthreads();
    }
    if (t < 16) {
        atomicAdd(&stats[(b * 16 + t) * 2 + 0], rs[t]);
        atomicAdd(&stats[(b * 16 + t) * 2 + 1], rq[t]);
    }
}

// ---------------- instance-norm apply + GELU (in place, f16) ----------------
__global__ void k_norm(f16* __restrict__ act, const float* __restrict__ stats, int npix) {
    size_t idx = (size_t)blockIdx.x * 256 + threadIdx.x;
    size_t total = (size_t)BB * npix * 16;
    if (idx >= total) return;
    int c = (int)(idx & 15);
    int b = (int)(idx / ((size_t)npix * 16));
    float s = stats[(b * 16 + c) * 2 + 0];
    float q = stats[(b * 16 + c) * 2 + 1];
    float inv = 1.f / (float)npix;
    float mean = s * inv;
    float var = q * inv - mean * mean;
    float r = rsqrtf(var + 1e-5f);
    float x = ((float)act[idx] - mean) * r;
    act[idx] = (f16)gelu_tanh(x);
}

// ---------------- last conv: 3x3, 16->2, tanh ----------------
__global__ void k_last(const f16* __restrict__ in, const float* __restrict__ w,
                       const float* __restrict__ bias, float* __restrict__ conved) {
    int idx = blockIdx.x * 256 + threadIdx.x;        // 16*32*32*2
    int co = idx & 1, x = (idx >> 1) & 31, y = (idx >> 6) & 31, b = idx >> 11;
    float acc = bias[co];
#pragma unroll
    for (int tap = 0; tap < 9; tap++) {
        int iy = y + tap / 3 - 1, ix = x + tap % 3 - 1;
        if (iy < 0 || iy >= 32 || ix < 0 || ix >= 32) continue;
        const f16* pin = in + (((size_t)b * 32 + iy) * 32 + ix) * 16;
#pragma unroll
        for (int ci = 0; ci < 16; ci++)
            acc += (float)pin[ci] * w[(tap * 16 + ci) * 2 + co];
    }
    conved[idx] = tanhf(acc);
}

// ---------------- cps = grid + 8 * pad(conved) -> d_out[1..] ----------------
__global__ void k_cps(const float* __restrict__ conved, float* __restrict__ outp) {
    int idx = blockIdx.x * 256 + threadIdx.x;        // 16*33*33*2 = 34848
    if (idx >= 16 * 33 * 33 * 2) return;
    int d = idx & 1;
    int tt = idx >> 1;
    int j = tt % 33, i = (tt / 33) % 33, b = tt / (33 * 33);
    float cv = (i < 32 && j < 32) ? conved[(((b * 32) + i) * 32 + j) * 2 + d] : 0.f;  // linear_ramp pad -> 0
    float g = (float)((d == 0 ? i : j) * 16);
    outp[1 + idx] = g + 8.f * cv;
}

// ---------------- area loss: one wave32 per (cfg, b, area) ----------------
__global__ void k_area(const float* __restrict__ img, const float* __restrict__ cpsbuf,
                       float* __restrict__ asum) {
    int wid = blockIdx.x * 8 + (threadIdx.x >> 5);   // 4*16*1024 = 65536 waves
    int lane = threadIdx.x & 31;
    int j = wid & 1023, b = (wid >> 10) & 15, cfg = wid >> 14;
    int sy = (cfg >> 1) * 8, sx = (cfg & 1) * 8;
    int by = j >> 5, bx = j & 31;
    float cpy = cpsbuf[1 + (((b * 33 + by) * 33 + bx) * 2) + 0];
    float cpx = cpsbuf[1 + (((b * 33 + by) * 33 + bx) * 2) + 1];
    float masked[8], mm[8];
#pragma unroll
    for (int t = 0; t < 8; t++) {
        int q = lane + 32 * t;
        int r = q >> 4, c = q & 15;
        int iy = sy + by * 16 + r - 8;
        int ix = sx + bx * 16 + c - 8;
        float iv = (iy >= 0 && iy < HH && ix >= 0 && ix < WW)
                       ? img[((size_t)b * HH + iy) * WW + ix] : 0.f;
        float dy = (float)iy - cpy, dx = (float)ix - cpx;
        float m = __expf(-(dy * dy + dx * dx) * (1.f / 128.f));
        m = diff_round(diff_round(m));
        mm[t] = m;
        masked[t] = iv * m;
    }
    float sm = 0.f, sk = 0.f;
#pragma unroll
    for (int t = 0; t < 8; t++) { sm += mm[t]; sk += masked[t]; }
    sm = wred(sm); sk = wred(sk);
    float msumT = sm + 1e-7f;
    float mean = (sk + 256.f * 1e-7f) / msumT;
    float sd = 0.f;
#pragma unroll
    for (int t = 0; t < 8; t++) {
        float dev = (masked[t] - mean) * mm[t];
        sd += dev * dev;
    }
    sd = wred(sd);
    if (lane == 0) atomicAdd(asum, (sd + 256.f * 1e-7f) / msumT);
}

// ---------------- edge sampling at cps ----------------
__global__ void k_eloss(const float* __restrict__ edge, const float* __restrict__ cpsbuf,
                        float* __restrict__ esum) {
    int idx = blockIdx.x * 256 + threadIdx.x;        // 16*1089
    if (idx >= 16 * 1089) return;
    int b = idx / 1089, p = idx % 1089;
    float yv = cpsbuf[1 + (b * 1089 + p) * 2 + 0];
    float xv = cpsbuf[1 + (b * 1089 + p) * 2 + 1];
    yv = fminf(fmaxf(yv, 0.f), 510.999f);
    xv = fminf(fmaxf(xv, 0.f), 510.999f);
    int y0 = (int)floorf(yv), x0 = (int)floorf(xv);
    float wy = yv - (float)y0, wx = xv - (float)x0;
    const float* e = edge + (size_t)b * HH * WW;
    float v00 = e[y0 * WW + x0],       v01 = e[y0 * WW + x0 + 1];
    float v10 = e[(y0 + 1) * WW + x0], v11 = e[(y0 + 1) * WW + x0 + 1];
    float v = v00 * (1 - wy) * (1 - wx) + v01 * (1 - wy) * wx
            + v10 * wy * (1 - wx)       + v11 * wy * wx;
    atomicAdd(esum, v);
}

__global__ void k_final(const float* __restrict__ scal, float* __restrict__ outp) {
    if (threadIdx.x == 0) {
        float aloss = scal[0] / (16.f * 4096.f);
        float eloss = scal[1] / (16.f * 1089.f);
        outp[0] = aloss * eloss;
    }
}

// ---------------- host ----------------
extern "C" void kernel_launch(void* const* d_in, const int* in_sizes, int n_in,
                              void* d_out, int out_size, void* d_ws, size_t ws_size,
                              hipStream_t stream) {
    (void)in_sizes; (void)n_in; (void)out_size; (void)ws_size;
    const float* image   = (const float*)d_in[0];
    const float* w_first = (const float*)d_in[1];
    const float* b_first = (const float*)d_in[2];
    const float* w_mid   = (const float*)d_in[3];
    const float* b_mid   = (const float*)d_in[4];
    const float* w_last  = (const float*)d_in[5];
    const float* b_last  = (const float*)d_in[6];
    float* outp = (float*)d_out;

    char* ws = (char*)d_ws;
    float* edge   = (float*)(ws + WS_EDGE);
    float* stats  = (float*)(ws + WS_STATS);
    float* scal   = (float*)(ws + WS_SCAL);
    float* conved = (float*)(ws + WS_CONVED);
    f16*   actA   = (f16*)(ws + WS_ACTA);
    f16*   actB   = (f16*)(ws + WS_ACTB);

    // zero stats (8*512) + scalars
    k_zero<<<17, 256, 0, stream>>>(stats, 4096 + 64);

    // edge map
    k_farid<<<(BB * HH * WW) / 256, 256, 0, stream>>>(image, edge);

    // layer 0: conv(2->16) @512^2, WMMA
    k_conv0<<<(BB * HH * (WW / 16)) / 8, 256, 0, stream>>>(image, edge, w_first, b_first, actA);
    {
        int npix = HH * WW, nch = 64;
        k_stats<<<BB * nch, 256, 0, stream>>>(actA, stats, npix, nch);
        size_t tot = (size_t)BB * npix * 16;
        k_norm<<<(unsigned)((tot + 255) / 256), 256, 0, stream>>>(actA, stats, npix);
    }

    // mid layers: strides 2,2,2,2,1,1,1
    f16* cur = actA;
    f16* nxt = actB;
    int Hi = 512;
    for (int i = 0; i < 7; i++) {
        int s = (i < 4) ? 2 : 1;
        int Ho = Hi / s;
        int tiles = BB * Ho * (Ho / 16);
        k_convmid<<<tiles / 8, 256, 0, stream>>>(cur, w_mid + i * 2304, b_mid + i * 16,
                                                 nxt, Hi, Hi, Ho, Ho, s);
        int npix = Ho * Ho;
        int nch = npix >= 4096 ? npix / 4096 : 1;
        float* st = stats + (1 + i) * 512;
        k_stats<<<BB * nch, 256, 0, stream>>>(nxt, st, npix, nch);
        size_t tot = (size_t)BB * npix * 16;
        k_norm<<<(unsigned)((tot + 255) / 256), 256, 0, stream>>>(nxt, st, npix);
        f16* tmp = cur; cur = nxt; nxt = tmp;
        Hi = Ho;
    }
    // after 7 layers: cur = actB holds 32x32x16

    k_last<<<(BB * 32 * 32 * 2) / 256, 256, 0, stream>>>(cur, w_last, b_last, conved);
    k_cps<<<(16 * 33 * 33 * 2 + 255) / 256, 256, 0, stream>>>(conved, outp);
    k_area<<<(4 * BB * 1024) / 8, 256, 0, stream>>>(image, outp, &scal[0]);
    k_eloss<<<(BB * 1089 + 255) / 256, 256, 0, stream>>>(edge, outp, &scal[1]);
    k_final<<<1, 32, 0, stream>>>(scal, outp);
}
